// Attention_53498112639067
// MI455X (gfx1250) — compile-verified, matrix operand order checked by reading, not run on hardware
//
#include <hip/hip_runtime.h>
#include <hip/hip_bf16.h>

#define B_    2
#define L_    2048
#define D_    1024
#define H_    16
#define HD_   64
#define NQKV_ 3072
#define MTOT_ (B_ * L_)   // 4096

typedef __bf16 bf16;
typedef __attribute__((ext_vector_type(16))) __bf16 v16bf;
typedef __attribute__((ext_vector_type(8)))  float   v8f;
typedef __attribute__((ext_vector_type(2)))  __bf16  bf16x2;

// Types matching the async-load builtin prototype "vV4i*1V4i*3IiIi":
// pointer to GCC-style int4 vector, in addrspace(1) / addrspace(3).
typedef int vi4 __attribute__((vector_size(16)));
typedef __attribute__((address_space(1))) vi4 as1_vi4;
typedef __attribute__((address_space(3))) vi4 as3_vi4;

// CDNA5 async global->LDS copies (ASYNCcnt path); guarded so that a toolchain
// without these builtins falls back to the VGPR round-trip staging path.
#if defined(__HIP_DEVICE_COMPILE__) && \
    __has_builtin(__builtin_amdgcn_global_load_async_to_lds_b128) && \
    __has_builtin(__builtin_amdgcn_s_wait_asynccnt)
#define HAVE_ASYNC_LDS 1
#else
#define HAVE_ASYNC_LDS 0
#endif

// ---------------------------------------------------------------- helpers

static __device__ __forceinline__ bf16 f2bf(float f) {
  union { float f; unsigned u; } c; c.f = f;
  unsigned r = c.u + 0x7FFFu + ((c.u >> 16) & 1u);   // round-to-nearest-even
  unsigned short h = (unsigned short)(r >> 16);
  return __builtin_bit_cast(bf16, h);
}

static __device__ __forceinline__ v8f zero8() {
  v8f z;
#pragma unroll
  for (int i = 0; i < 8; ++i) z[i] = 0.0f;
  return z;
}

// Copy 16 bytes global -> LDS. Async (no VGPR data round-trip) when available.
static __device__ __forceinline__ void stage_b128(const bf16* g, bf16* l) {
#if HAVE_ASYNC_LDS
  __builtin_amdgcn_global_load_async_to_lds_b128(
      (as1_vi4*)(unsigned long long)g,
      (as3_vi4*)(unsigned int)(unsigned long long)l,
      0, 0);
#else
  *(uint4*)l = *(const uint4*)g;
#endif
}

static __device__ __forceinline__ void wait_stage() {
#if HAVE_ASYNC_LDS
  __builtin_amdgcn_s_wait_asynccnt(0);
#endif
}

// 16x32 bf16 WMMA fragment (ISA 16-bit A layout; identical for B when the
// source tile is stored "row = N, col = K", i.e. K-major per row).
// lane%16 = M (or N) row; VGPR r holds K = 2r + 8*(lane/16), +8 when r>=4.
static __device__ __forceinline__ v16bf ld_frag(const bf16* p, int ld) {
  const int lane = threadIdx.x & 31;
  const int rn = lane & 15;
  const int hf = lane >> 4;
  v16bf f;
#pragma unroll
  for (int r = 0; r < 8; ++r) {
    const int kb = 2 * r + 8 * hf + ((r & 4) ? 8 : 0);
    bf16x2 p2 = *(const bf16x2*)(p + rn * ld + kb);
    f[2 * r]     = p2.x;
    f[2 * r + 1] = p2.y;
  }
  return f;
}

static __device__ __forceinline__ v8f wmma_bf16(v16bf a, v16bf b, v8f c) {
  return __builtin_amdgcn_wmma_f32_16x16x32_bf16(
      false, a, false, b, (short)0, c, false, false);
}

// ---------------------------------------------------------------- kernels

__global__ void cvt_f32_bf16(const float* __restrict__ src,
                             bf16* __restrict__ dst, int n) {
  int i = blockIdx.x * blockDim.x + threadIdx.x;
  if (i < n) dst[i] = f2bf(src[i]);
}

// C[M,N] = A[M,K] * B[K,N] (+bias), bf16 inputs, fp32 out.
// Block: 256 threads = 8 waves; block tile 128x128; wave tile 32x64
// (2 A-frags x 4 B-frags = 8 WMMA per wave per 32-deep K step).
__global__ __launch_bounds__(256)
void gemm_bf16(const bf16* __restrict__ A, const bf16* __restrict__ Bm,
               float* __restrict__ C, const float* __restrict__ bias,
               int M, int N, int K) {
  __shared__ __align__(16) bf16 As[128 * 32];    // [m][k]
  __shared__ __align__(16) bf16 BsT[128 * 32];   // [n][k]

  const int tid  = threadIdx.x;
  const int lane = tid & 31;
  const int wid  = tid >> 5;
  const int wm   = wid & 3;     // 32-row tile within block
  const int wn   = wid >> 2;    // 64-col slab within block
  const int m0   = blockIdx.x * 128;
  const int n0   = blockIdx.y * 128;

  v8f acc[2][4];
#pragma unroll
  for (int mt = 0; mt < 2; ++mt)
#pragma unroll
    for (int nt = 0; nt < 4; ++nt) acc[mt][nt] = zero8();

  const int brow = tid >> 3, bcg = tid & 7;   // B stage: 32 rows x 8 groups of 16

  for (int k0 = 0; k0 < K; k0 += 32) {
    // stage A tile 128x32 (two b128 chunks per thread, async -> LDS)
#pragma unroll
    for (int c = 0; c < 2; ++c) {
      const int ch  = tid + 256 * c;
      const int row = ch >> 2, cg = ch & 3;
      stage_b128(A + (size_t)(m0 + row) * K + k0 + cg * 8,
                 As + row * 32 + cg * 8);
    }
    // stage B tile 32x128 transposed -> BsT[n][k]
    {
      const bf16* g = Bm + (size_t)(k0 + brow) * N + n0 + bcg * 16;
      union { uint4 u; bf16 e[8]; } t0, t1;
      t0.u = *(const uint4*)g;
      t1.u = *(const uint4*)(g + 8);
#pragma unroll
      for (int j = 0; j < 8; ++j) {
        BsT[(bcg * 16 + j) * 32 + brow]     = t0.e[j];
        BsT[(bcg * 16 + 8 + j) * 32 + brow] = t1.e[j];
      }
    }
    if (k0 + 32 < K) {
      __builtin_prefetch(A + (size_t)(m0 + tid) * K + (k0 + 32), 0, 3);
      __builtin_prefetch(Bm + (size_t)(k0 + 32 + brow) * N + n0 + bcg * 16, 0, 3);
    }
    wait_stage();
    __syncthreads();

    v16bf af[2];
#pragma unroll
    for (int mt = 0; mt < 2; ++mt)
      af[mt] = ld_frag(As + (wm * 32 + mt * 16) * 32, 32);
#pragma unroll
    for (int nt = 0; nt < 4; ++nt) {
      const v16bf bfr = ld_frag(BsT + (wn * 64 + nt * 16) * 32, 32);
#pragma unroll
      for (int mt = 0; mt < 2; ++mt)
        acc[mt][nt] = wmma_bf16(af[mt], bfr, acc[mt][nt]);
    }
    __syncthreads();
  }

  const int nn = lane & 15, hf = lane >> 4;
#pragma unroll
  for (int mt = 0; mt < 2; ++mt)
#pragma unroll
    for (int nt = 0; nt < 4; ++nt) {
      const int col = n0 + wn * 64 + nt * 16 + nn;
      const float bv = bias ? bias[col] : 0.0f;
#pragma unroll
      for (int r = 0; r < 8; ++r) {
        const int row = m0 + wm * 32 + mt * 16 + r + 8 * hf;
        C[(size_t)row * N + col] = acc[mt][nt][r] + bv;
      }
    }
}

// RoPE on q,k; split fused qkv (fp32 [B,L,3072]) into bf16 q/k/v [B,H,L,64].
__global__ void rope_split(const float* __restrict__ qkv,
                           const float* __restrict__ cosT,
                           const float* __restrict__ sinT,
                           bf16* __restrict__ qo, bf16* __restrict__ ko,
                           bf16* __restrict__ vo) {
  int idx = blockIdx.x * blockDim.x + threadIdx.x;   // over B*L*H*HD
  if (idx >= B_ * L_ * H_ * HD_) return;
  const int d = idx & 63;
  const int h = (idx >> 6) & 15;
  const int l = (idx >> 10) & 2047;
  const int b = idx >> 21;
  const size_t rbase = (size_t)(b * L_ + l) * NQKV_ + h * HD_;
  const float c = cosT[l * HD_ + d];
  const float s = sinT[l * HD_ + d];
  const int   d2  = (d < 32) ? d + 32 : d - 32;
  const float sgn = (d < 32) ? -1.0f : 1.0f;
  const float q  = qkv[rbase + d],        q2 = qkv[rbase + d2];
  const float k  = qkv[rbase + 1024 + d], k2 = qkv[rbase + 1024 + d2];
  const float v  = qkv[rbase + 2048 + d];
  const size_t ob = (size_t)((b * H_ + h) * L_ + l) * HD_ + d;
  qo[ob] = f2bf(q * c + sgn * q2 * s);
  ko[ob] = f2bf(k * c + sgn * k2 * s);
  vo[ob] = f2bf(v);
}

// Causal flash attention. Block = (b,h) x 128 queries; wave = 16 queries.
__global__ __launch_bounds__(256)
void attn_fwd(const bf16* __restrict__ qm, const bf16* __restrict__ km,
              const bf16* __restrict__ vm, bf16* __restrict__ om) {
  __shared__ __align__(16) bf16 Ks[32 * 64];        // [key][d]
  __shared__ __align__(16) bf16 VsT[64 * 32];       // [d][key]
  __shared__ __align__(16) bf16 Ps[8 * 16 * 32];    // per-wave P [q][key]

  const int tid = threadIdx.x, lane = tid & 31, wid = tid >> 5;
  const int nn = lane & 15, hf = lane >> 4;

  const int bh   = blockIdx.x >> 4;        // 16 q-blocks per (b,h)
  const int qblk = blockIdx.x & 15;
  const int q0   = qblk * 128;
  const int qrow0 = q0 + wid * 16;
  const size_t base = (size_t)bh * L_ * HD_;
  const int b = bh >> 4, h = bh & 15;

  const v16bf aq0 = ld_frag(qm + base + (size_t)qrow0 * HD_ + 0,  HD_);
  const v16bf aq1 = ld_frag(qm + base + (size_t)qrow0 * HD_ + 32, HD_);

  float mrow[8], lrow[8];
  v8f oacc[4];
#pragma unroll
  for (int r = 0; r < 8; ++r) { mrow[r] = -3.0e38f; lrow[r] = 0.0f; }
#pragma unroll
  for (int t = 0; t < 4; ++t) oacc[t] = zero8();

  const int krow = tid >> 3, kcg = tid & 7;
  const int nkt = (q0 + 128) / 32;
  const float scale = 0.125f;   // 1/sqrt(64)

  for (int kt = 0; kt < nkt; ++kt) {
    const int k0 = kt * 32;
    // stage K tile [32 x 64] (async -> LDS when available)
    stage_b128(km + base + (size_t)(k0 + krow) * HD_ + kcg * 8,
               Ks + krow * 64 + kcg * 8);
    // stage V tile transposed -> VsT[d][key]
    {
      union { uint4 u; bf16 e[8]; } tv;
      tv.u = *(const uint4*)(vm + base + (size_t)(k0 + krow) * HD_ + kcg * 8);
#pragma unroll
      for (int j = 0; j < 8; ++j) VsT[(kcg * 8 + j) * 32 + krow] = tv.e[j];
    }
    wait_stage();
    __syncthreads();

    if (k0 < qrow0 + 16) {    // this wave has unmasked keys in the tile
      v8f s0 = zero8(), s1 = zero8();
      {
        v16bf bk = ld_frag(Ks + 0 * 64 + 0, 64);  s0 = wmma_bf16(aq0, bk, s0);
        bk = ld_frag(Ks + 0 * 64 + 32, 64);       s0 = wmma_bf16(aq1, bk, s0);
        bk = ld_frag(Ks + 16 * 64 + 0, 64);       s1 = wmma_bf16(aq0, bk, s1);
        bk = ld_frag(Ks + 16 * 64 + 32, 64);      s1 = wmma_bf16(aq1, bk, s1);
      }
      bf16* Pw = Ps + wid * (16 * 32);
#pragma unroll
      for (int r = 0; r < 8; ++r) {
        const int qi = qrow0 + r + 8 * hf;
        float a0 = s0[r] * scale, a1 = s1[r] * scale;
        if (k0 + nn > qi)      a0 = -3.0e38f;        // causal mask
        if (k0 + 16 + nn > qi) a1 = -3.0e38f;
        float rm = fmaxf(a0, a1);
#pragma unroll
        for (int off = 8; off > 0; off >>= 1)
          rm = fmaxf(rm, __shfl_xor(rm, off, 16));
        const float mnew = fmaxf(mrow[r], rm);
        const float corr = __expf(mrow[r] - mnew);
        const float p0 = __expf(a0 - mnew);
        const float p1 = __expf(a1 - mnew);
        float ps = p0 + p1;
#pragma unroll
        for (int off = 8; off > 0; off >>= 1)
          ps += __shfl_xor(ps, off, 16);
        lrow[r] = lrow[r] * corr + ps;
        mrow[r] = mnew;
#pragma unroll
        for (int t = 0; t < 4; ++t) oacc[t][r] *= corr;
        Pw[(r + 8 * hf) * 32 + nn]      = f2bf(p0);  // re-lay P for A-fragment
        Pw[(r + 8 * hf) * 32 + 16 + nn] = f2bf(p1);
      }
      const v16bf ap = ld_frag(Pw, 32);
#pragma unroll
      for (int nt = 0; nt < 4; ++nt) {
        const v16bf bv = ld_frag(VsT + (nt * 16) * 32, 32);
        oacc[nt] = wmma_bf16(ap, bv, oacc[nt]);
      }
    }
    __syncthreads();
  }

  // write O (bf16) into [B, L, H*HD] for the output projection
#pragma unroll
  for (int r = 0; r < 8; ++r) {
    const int qi = qrow0 + r + 8 * hf;
    const float inv = 1.0f / lrow[r];
    bf16* orow = om + (size_t)(b * L_ + qi) * (H_ * HD_) + h * HD_;
#pragma unroll
    for (int nt = 0; nt < 4; ++nt)
      orow[nt * 16 + nn] = f2bf(oacc[nt][r] * inv);
  }
}

// ---------------------------------------------------------------- launch

extern "C" void kernel_launch(void* const* d_in, const int* in_sizes, int n_in,
                              void* d_out, int out_size, void* d_ws, size_t ws_size,
                              hipStream_t stream) {
  const float* x    = (const float*)d_in[0];
  const float* cosT = (const float*)d_in[1];
  const float* sinT = (const float*)d_in[2];
  const float* Wqkv = (const float*)d_in[3];
  const float* Wout = (const float*)d_in[4];
  const float* bout = (const float*)d_in[5];
  float* out = (float*)d_out;

  char* w = (char*)d_ws;
  auto alloc = [&](size_t bytes) {
    char* p = w;
    w += (bytes + 255) & ~(size_t)255;
    return p;
  };
  bf16*  xb    = (bf16*)alloc((size_t)MTOT_ * D_ * 2);
  bf16*  wqkvb = (bf16*)alloc((size_t)D_ * NQKV_ * 2);
  bf16*  woutb = (bf16*)alloc((size_t)D_ * D_ * 2);
  float* qkv   = (float*)alloc((size_t)MTOT_ * NQKV_ * 4);
  bf16*  qb    = (bf16*)alloc((size_t)MTOT_ * D_ * 2);
  bf16*  kb    = (bf16*)alloc((size_t)MTOT_ * D_ * 2);
  bf16*  vb    = (bf16*)alloc((size_t)MTOT_ * D_ * 2);
  bf16*  ob    = (bf16*)alloc((size_t)MTOT_ * D_ * 2);

  cvt_f32_bf16<<<(MTOT_ * D_) / 256, 256, 0, stream>>>(x, xb, MTOT_ * D_);
  cvt_f32_bf16<<<(D_ * NQKV_) / 256, 256, 0, stream>>>(Wqkv, wqkvb, D_ * NQKV_);
  cvt_f32_bf16<<<(D_ * D_) / 256, 256, 0, stream>>>(Wout, woutb, D_ * D_);

  // QKV projection: [4096,1024] x [1024,3072] -> fp32
  gemm_bf16<<<dim3(MTOT_ / 128, NQKV_ / 128), 256, 0, stream>>>(
      xb, wqkvb, qkv, nullptr, MTOT_, NQKV_, D_);

  rope_split<<<(B_ * L_ * H_ * HD_) / 256, 256, 0, stream>>>(
      qkv, cosT, sinT, qb, kb, vb);

  attn_fwd<<<B_ * H_ * (L_ / 128), 256, 0, stream>>>(qb, kb, vb, ob);

  // Output projection + bias: [4096,1024] x [1024,1024] -> d_out fp32
  gemm_bf16<<<dim3(MTOT_ / 128, D_ / 128), 256, 0, stream>>>(
      ob, woutb, out, bout, MTOT_, D_, D_);
}
